// GPT_86474871538359
// MI455X (gfx1250) — matrix operand order repaired and checked by reading, not compiled, not run on hardware
//
#include <hip/hip_runtime.h>
#include <math.h>

typedef __attribute__((ext_vector_type(16))) __bf16 v16bf;
typedef __attribute__((ext_vector_type(8)))  float  v8f;
typedef __attribute__((ext_vector_type(4)))  int    v4i;

#define WMMA_BF16(a, b, c) \
  __builtin_amdgcn_wmma_f32_16x16x32_bf16(false, (a), false, (b), (short)0, (c), false, false)

// gfx1250 async global->LDS path (ASYNCcnt), with compile-safe fallback.
#if defined(__has_builtin)
#if __has_builtin(__builtin_amdgcn_global_load_async_to_lds_b128) && \
    __has_builtin(__builtin_amdgcn_s_wait_asynccnt)
#define USE_ASYNC_LDS 1
#endif
#endif
#ifndef USE_ASYNC_LDS
#define USE_ASYNC_LDS 0
#endif

#if USE_ASYNC_LDS
typedef __attribute__((address_space(1))) v4i GV4;   // global (prints as __device__)
typedef __attribute__((address_space(3))) v4i LV4;   // LDS (prints as __shared__)
__device__ __forceinline__ void async_g2l_b128(const uint4* g, uint4* l) {
  __builtin_amdgcn_global_load_async_to_lds_b128((GV4*)g, (LV4*)l, 0, 0);
}
#endif

union FragBF { v16bf v; unsigned int u[8]; };

__device__ __forceinline__ unsigned short f2b(float f) {
  unsigned int u = __float_as_uint(f);
  u += 0x7FFFu + ((u >> 16) & 1u);          // round-to-nearest-even
  return (unsigned short)(u >> 16);
}

// ---------------------------------------------------------------- cvt fp32->bf16
__global__ __launch_bounds__(256) void cvt_f32_bf16(const float* __restrict__ in,
                                                    unsigned short* __restrict__ out,
                                                    size_t n) {
  size_t i = (size_t)blockIdx.x * blockDim.x + threadIdx.x;
  size_t st = (size_t)gridDim.x * blockDim.x;
  for (; i < n; i += st) out[i] = f2b(in[i]);
}

// ---------------------------------------------------------------- x = seq + pos
__global__ __launch_bounds__(256) void add_pos(const float* __restrict__ seq,
                                               const float* __restrict__ pos,
                                               float* __restrict__ x,
                                               size_t n, size_t tc) {
  size_t i = (size_t)blockIdx.x * blockDim.x + threadIdx.x;
  size_t st = (size_t)gridDim.x * blockDim.x;
  for (; i < n; i += st) x[i] = seq[i] + pos[i % tc];
}

// ---------------------------------------------------------------- LayerNorm -> bf16
__global__ __launch_bounds__(256) void layernorm_bf16(const float* __restrict__ x,
                                                      const float* __restrict__ w,
                                                      const float* __restrict__ b,
                                                      unsigned short* __restrict__ out,
                                                      int Cn) {
  int row = blockIdx.x;
  const float* xr = x + (size_t)row * Cn;
  float s = 0.f, s2 = 0.f;
  for (int c = threadIdx.x; c < Cn; c += 256) { float v = xr[c]; s += v; s2 += v * v; }
  __shared__ float rA[8], rB[8];
  __shared__ float mu_s, rs_s;
  for (int o = 16; o; o >>= 1) { s += __shfl_xor(s, o, 32); s2 += __shfl_xor(s2, o, 32); }
  int lane = threadIdx.x & 31, wv = threadIdx.x >> 5;
  if (lane == 0) { rA[wv] = s; rB[wv] = s2; }
  __syncthreads();
  if (threadIdx.x == 0) {
    float ts = 0.f, ts2 = 0.f;
    for (int i = 0; i < 8; ++i) { ts += rA[i]; ts2 += rB[i]; }
    float mu = ts / (float)Cn;
    float var = ts2 / (float)Cn - mu * mu;
    mu_s = mu; rs_s = rsqrtf(var + 1e-5f);
  }
  __syncthreads();
  float mu = mu_s, rs = rs_s;
  for (int c = threadIdx.x; c < Cn; c += 256)
    out[(size_t)row * Cn + c] = f2b((xr[c] - mu) * rs * w[c] + b[c]);
}

// ---------------------------------------------------------------- generic bf16 WMMA GEMM
// D[M,N] = act(A[M,K] @ W[N,K]^T [+ bias]) [+ resid]; out fp32 (OUTF) or bf16.
// Block tile 128x64, K-step 32, double-buffered LDS pipeline (1 barrier/step).
// 4 waves, each wave owns 32 rows (2 M-subtiles), 8 WMMA per step per wave.
// M multiple of 128; N multiple of 64; K multiple of 32.
template <bool HAS_BIAS, bool GELU, bool RESID, bool OUTF>
__global__ __launch_bounds__(128) void gemm_bf16(const unsigned short* __restrict__ A,
                                                 const unsigned short* __restrict__ W,
                                                 const float* __restrict__ bias,
                                                 const float* __restrict__ resid,
                                                 float* __restrict__ outF,
                                                 unsigned short* __restrict__ outB,
                                                 int M, int N, int K) {
  __shared__ uint4 As4[2][128 * 4];   // 128 rows x 32 cols bf16 per buffer
  __shared__ uint4 Ws4[2][64 * 4];    // 64 rows x 32 cols bf16 per buffer
  const int bm = blockIdx.y * 128, bn = blockIdx.x * 64;
  const int tid = threadIdx.x, lane = tid & 31, wave = tid >> 5;
  const int m16 = lane & 15, hf = lane >> 4;
  const int Kp4 = K >> 3;                        // row stride in uint4
  const uint4* Ag4 = (const uint4*)A;
  const uint4* Wg4 = (const uint4*)W;
  const int steps = K >> 5;
  v8f acc[2][4] = {};

  // ---- stage tile 0 into buffer 0
#if USE_ASYNC_LDS
#pragma unroll
  for (int i = 0; i < 4; ++i) {
    int p = tid + i * 128, r = p >> 2, q = p & 3;
    async_g2l_b128(&Ag4[(size_t)(bm + r) * Kp4 + q], &As4[0][p]);
  }
#pragma unroll
  for (int i = 0; i < 2; ++i) {
    int p = tid + i * 128, r = p >> 2, q = p & 3;
    async_g2l_b128(&Wg4[(size_t)(bn + r) * Kp4 + q], &Ws4[0][p]);
  }
  __builtin_amdgcn_s_wait_asynccnt(0);
#else
  {
    uint4 aR[4], wR[2];
#pragma unroll
    for (int i = 0; i < 4; ++i) {
      int p = tid + i * 128, r = p >> 2, q = p & 3;
      aR[i] = Ag4[(size_t)(bm + r) * Kp4 + q];
    }
#pragma unroll
    for (int i = 0; i < 2; ++i) {
      int p = tid + i * 128, r = p >> 2, q = p & 3;
      wR[i] = Wg4[(size_t)(bn + r) * Kp4 + q];
    }
#pragma unroll
    for (int i = 0; i < 4; ++i) As4[0][tid + i * 128] = aR[i];
#pragma unroll
    for (int i = 0; i < 2; ++i) Ws4[0][tid + i * 128] = wR[i];
  }
#endif
  __syncthreads();

  for (int t = 0; t < steps; ++t) {
    const int cur = t & 1, nxt = cur ^ 1;
    const int kq = (t + 1) << 2;                 // next tile offset in uint4
    const bool haveNext = (t + 1 < steps);
#if USE_ASYNC_LDS
    if (haveNext) {                              // async-issue next tile, overlap with WMMA
#pragma unroll
      for (int i = 0; i < 4; ++i) {
        int p = tid + i * 128, r = p >> 2, q = p & 3;
        async_g2l_b128(&Ag4[(size_t)(bm + r) * Kp4 + kq + q], &As4[nxt][p]);
      }
#pragma unroll
      for (int i = 0; i < 2; ++i) {
        int p = tid + i * 128, r = p >> 2, q = p & 3;
        async_g2l_b128(&Wg4[(size_t)(bn + r) * Kp4 + kq + q], &Ws4[nxt][p]);
      }
    }
#else
    uint4 aR[4], wR[2];
    if (haveNext) {                              // issue next-tile loads, overlap with WMMA
#pragma unroll
      for (int i = 0; i < 4; ++i) {
        int p = tid + i * 128, r = p >> 2, q = p & 3;
        aR[i] = Ag4[(size_t)(bm + r) * Kp4 + kq + q];
      }
#pragma unroll
      for (int i = 0; i < 2; ++i) {
        int p = tid + i * 128, r = p >> 2, q = p & 3;
        wR[i] = Wg4[(size_t)(bn + r) * Kp4 + kq + q];
      }
    }
#endif
    // ---- compute on current buffer
    const unsigned int* As = (const unsigned int*)As4[cur];
    const unsigned int* Ws = (const unsigned int*)Ws4[cur];
    FragBF af[2];
#pragma unroll
    for (int mi = 0; mi < 2; ++mi)
#pragma unroll
      for (int vv = 0; vv < 8; ++vv) {
        int kk = ((vv < 4) ? 0 : 16) + hf * 8 + (vv & 3) * 2;   // A layout
        af[mi].u[vv] = As[(wave * 32 + mi * 16 + m16) * 16 + (kk >> 1)];
      }
#pragma unroll
    for (int j = 0; j < 4; ++j) {
      FragBF bf;
#pragma unroll
      for (int vv = 0; vv < 8; ++vv)             // B layout: col = m16, K pair = (vv<4?0:8)+hf*4+(vv&3)
        bf.u[vv] = Ws[(j * 16 + m16) * 16 + (((vv < 4) ? 0 : 8) + hf * 4 + (vv & 3))];
      acc[0][j] = WMMA_BF16(af[0].v, bf.v, acc[0][j]);
      acc[1][j] = WMMA_BF16(af[1].v, bf.v, acc[1][j]);
    }
#if !USE_ASYNC_LDS
    if (haveNext) {                              // commit next tile to other buffer
#pragma unroll
      for (int i = 0; i < 4; ++i) As4[nxt][tid + i * 128] = aR[i];
#pragma unroll
      for (int i = 0; i < 2; ++i) Ws4[nxt][tid + i * 128] = wR[i];
    }
#endif
    if (haveNext) {
#if USE_ASYNC_LDS
      __builtin_amdgcn_s_wait_asynccnt(0);
#endif
      __syncthreads();                           // single barrier per step
    }
  }

#pragma unroll
  for (int mi = 0; mi < 2; ++mi)
#pragma unroll
    for (int j = 0; j < 4; ++j) {
      int n = bn + j * 16 + m16;
      float bv = HAS_BIAS ? bias[n] : 0.0f;
#pragma unroll
      for (int r = 0; r < 8; ++r) {
        int m = bm + wave * 32 + mi * 16 + r + hf * 8;
        float val = acc[mi][j][r] + bv;
        if (GELU) val = 0.5f * val * (1.0f + erff(val * 0.70710678118654752f));
        size_t idx = (size_t)m * N + n;
        if (RESID) val += resid[idx];
        if (OUTF) outF[idx] = val;
        else      outB[idx] = f2b(val);
      }
    }
}

// ---------------------------------------------------------------- flash attention
// q,k,v,y layout: [B,T,C] viewed as [B,T,H,DH]; block = 64 queries for one (b,h)
__global__ __launch_bounds__(128) void flash_attn(const unsigned short* __restrict__ Q,
                                                  const unsigned short* __restrict__ Km,
                                                  const unsigned short* __restrict__ V,
                                                  unsigned short* __restrict__ Y) {
  const int Tn = 2048, Cn = 1024, DHn = 64;
  const int qt = blockIdx.x;
  const int bh = blockIdx.y;
  const int bb = bh >> 4, hh = bh & 15;
  const int tid = threadIdx.x, lane = tid & 31, wave = tid >> 5;
  const int m16 = lane & 15, hf = lane >> 4;
  __shared__ uint4 Qs4[64 * 8];            // 64 q rows x 64 d (8 uint4/row)
  __shared__ uint4 Ks4[32 * 8];            // 32 keys x 64 d
  __shared__ unsigned int Vs[64 * 16];     // 64 d x 32 keys (key pairs, transposed)
  __shared__ unsigned int Ps[4][256];      // per-wave 16x32 probs (pairs)
  const unsigned int* Qs = (const unsigned int*)Qs4;
  const unsigned int* Ks = (const unsigned int*)Ks4;
  unsigned short* VsU = (unsigned short*)Vs;
  const size_t base = ((size_t)bb * Tn) * Cn + (size_t)hh * DHn;   // u16 units
  const size_t gbase = base >> 1;                                  // u32 units
  const size_t gbase4 = base >> 3;                                 // uint4 units
  const uint4* Qg4 = (const uint4*)Q;
  const uint4* Kg4 = (const uint4*)Km;
  const unsigned int* Vg = (const unsigned int*)V;

#pragma unroll
  for (int i = 0; i < 4; ++i) {            // Q tile: 512 uint4
    int p = tid + i * 128;
    int r = p >> 3, q = p & 7;
    Qs4[p] = Qg4[gbase4 + (size_t)(qt * 64 + r) * (Cn / 8) + q];
  }
  __syncthreads();

  FragBF qf[2];
#pragma unroll
  for (int c = 0; c < 2; ++c)
#pragma unroll
    for (int vv = 0; vv < 8; ++vv) {
      int kk = c * 32 + ((vv < 4) ? 0 : 16) + hf * 8 + (vv & 3) * 2;
      qf[c].u[vv] = Qs[(wave * 16 + m16) * 32 + (kk >> 1)];
    }

  float rowm[8], rowl[8];
#pragma unroll
  for (int r = 0; r < 8; ++r) { rowm[r] = -1e30f; rowl[r] = 0.f; }
  v8f o[4] = {};

  for (int kt = 0; kt < Tn / 32; ++kt) {
    __syncthreads();
#pragma unroll
    for (int i = 0; i < 2; ++i) {          // K tile: 256 uint4
      int p = tid + i * 128;
      int r = p >> 3, q = p & 7;
      Ks4[p] = Kg4[gbase4 + (size_t)(kt * 32 + r) * (Cn / 8) + q];
    }
#pragma unroll
    for (int i = 0; i < 8; ++i) {          // V tile: read b32, scatter-transpose to LDS
      int p = tid + i * 128;               // 0..1023: key = p>>5, dpair = p&31
      int key = p >> 5, dp = p & 31;
      unsigned int w = Vg[gbase + (size_t)(kt * 32 + key) * (Cn / 2) + dp];
      VsU[(2 * dp) * 32 + key]     = (unsigned short)(w & 0xFFFFu);
      VsU[(2 * dp + 1) * 32 + key] = (unsigned short)(w >> 16);
    }
    if (kt + 1 < Tn / 32) {                // prefetch next K/V tiles
      __builtin_prefetch(&Kg4[gbase4 + (size_t)((kt + 1) * 32 + (tid >> 3)) * (Cn / 8) + (tid & 7)], 0, 1);
      __builtin_prefetch(&Vg[gbase + (size_t)((kt + 1) * 32 + (tid >> 2)) * (Cn / 2) + (tid & 3) * 8], 0, 1);
    }
    __syncthreads();

    // S = Q @ K^T : two 16x16 accumulators (key halves)
    v8f s0 = {}, s1 = {};
#pragma unroll
    for (int c = 0; c < 2; ++c) {
      FragBF kf0, kf1;
#pragma unroll
      for (int vv = 0; vv < 8; ++vv) {
        int pr = c * 16 + ((vv < 4) ? 0 : 8) + hf * 4 + (vv & 3);     // d pair index (B layout)
        kf0.u[vv] = Ks[(m16)*32 + pr];
        kf1.u[vv] = Ks[(16 + m16) * 32 + pr];
      }
      s0 = WMMA_BF16(qf[c].v, kf0.v, s0);
      s1 = WMMA_BF16(qf[c].v, kf1.v, s1);
    }

    // online softmax (per-row stats replicated across the 16 lanes of each half)
    float p0[8], p1[8], rscale[8];
#pragma unroll
    for (int r = 0; r < 8; ++r) {
      float a = s0[r] * 0.125f, b = s1[r] * 0.125f;   // 1/sqrt(64)
      float mx = fmaxf(a, b);
      mx = fmaxf(mx, __shfl_xor(mx, 1, 32));
      mx = fmaxf(mx, __shfl_xor(mx, 2, 32));
      mx = fmaxf(mx, __shfl_xor(mx, 4, 32));
      mx = fmaxf(mx, __shfl_xor(mx, 8, 32));
      float newm = fmaxf(rowm[r], mx);
      float sc = __expf(rowm[r] - newm);
      float e0 = __expf(a - newm), e1 = __expf(b - newm);
      float rsum = e0 + e1;
      rsum += __shfl_xor(rsum, 1, 32);
      rsum += __shfl_xor(rsum, 2, 32);
      rsum += __shfl_xor(rsum, 4, 32);
      rsum += __shfl_xor(rsum, 8, 32);
      rowl[r] = rowl[r] * sc + rsum;
      rowm[r] = newm;
      rscale[r] = sc;
      p0[r] = e0; p1[r] = e1;
    }

    // C-layout -> A-layout bounce through private per-wave LDS (wave lockstep)
    unsigned short* PsU = (unsigned short*)Ps[wave];
#pragma unroll
    for (int r = 0; r < 8; ++r) {
      PsU[(r + 8 * hf) * 32 + m16]      = f2b(p0[r]);
      PsU[(r + 8 * hf) * 32 + 16 + m16] = f2b(p1[r]);
    }
    FragBF pf;
#pragma unroll
    for (int vv = 0; vv < 8; ++vv) {
      int kk = ((vv < 4) ? 0 : 16) + hf * 8 + (vv & 3) * 2;
      pf.u[vv] = Ps[wave][m16 * 16 + (kk >> 1)];
    }

    // O = O*scale + P @ V
#pragma unroll
    for (int j = 0; j < 4; ++j) {
      FragBF vf;
#pragma unroll
      for (int vv = 0; vv < 8; ++vv) {
        int d = j * 16 + m16;
        vf.u[vv] = Vs[d * 16 + (((vv < 4) ? 0 : 8) + hf * 4 + (vv & 3))];   // key-pair index (B layout)
      }
#pragma unroll
      for (int r = 0; r < 8; ++r) o[j][r] *= rscale[r];
      o[j] = WMMA_BF16(pf.v, vf.v, o[j]);
    }
  }

  float rinv[8];
#pragma unroll
  for (int r = 0; r < 8; ++r) rinv[r] = 1.0f / rowl[r];
#pragma unroll
  for (int j = 0; j < 4; ++j)
#pragma unroll
    for (int r = 0; r < 8; ++r) {
      int m = qt * 64 + wave * 16 + r + 8 * hf;
      int d = j * 16 + m16;
      Y[base + (size_t)m * Cn + d] = f2b(o[j][r] * rinv[r]);
    }
}

// ---------------------------------------------------------------- host launcher
extern "C" void kernel_launch(void* const* d_in, const int* in_sizes, int n_in,
                              void* d_out, int out_size, void* d_ws, size_t ws_size,
                              hipStream_t stream) {
  (void)in_sizes; (void)n_in; (void)out_size; (void)ws_size;
  const int Bc = 4, Tc = 2048, Cc = 1024, Lc = 8, FF = 4096, OUTc = 1024, Hc = 16;
  const size_t M = (size_t)Bc * Tc;   // 8192

  const float* seq  = (const float*)d_in[0];
  const float* pos  = (const float*)d_in[1];
  const float* Wq   = (const float*)d_in[2];
  const float* bq   = (const float*)d_in[3];
  const float* Wk   = (const float*)d_in[4];
  const float* bk   = (const float*)d_in[5];
  const float* Wv   = (const float*)d_in[6];
  const float* bv   = (const float*)d_in[7];
  const float* Wo   = (const float*)d_in[8];
  const float* bo   = (const float*)d_in[9];
  const float* l1w  = (const float*)d_in[10];
  const float* l1b  = (const float*)d_in[11];
  const float* l2w  = (const float*)d_in[12];
  const float* l2b  = (const float*)d_in[13];
  const float* W1   = (const float*)d_in[14];
  const float* b1   = (const float*)d_in[15];
  const float* W2   = (const float*)d_in[16];
  const float* b2   = (const float*)d_in[17];
  const float* lnfw = (const float*)d_in[18];
  const float* lnfb = (const float*)d_in[19];
  const float* Wh   = (const float*)d_in[20];

  // workspace carve-out (256B aligned)
  char* p = (char*)d_ws;
  auto carve = [&](size_t bytes) -> char* {
    char* r = p; p += (bytes + 255) & ~(size_t)255; return r;
  };
  const size_t nQK = (size_t)Lc * Cc * Cc;       // per q/k/v/o weight stack
  const size_t nFF = (size_t)Lc * FF * Cc;       // per W1/W2 stack
  unsigned short* WqB = (unsigned short*)carve(nQK * 2);
  unsigned short* WkB = (unsigned short*)carve(nQK * 2);
  unsigned short* WvB = (unsigned short*)carve(nQK * 2);
  unsigned short* WoB = (unsigned short*)carve(nQK * 2);
  unsigned short* W1B = (unsigned short*)carve(nFF * 2);
  unsigned short* W2B = (unsigned short*)carve(nFF * 2);
  unsigned short* WhB = (unsigned short*)carve((size_t)OUTc * Cc * 2);
  float*          x   = (float*)carve(M * Cc * 4);
  unsigned short* hb  = (unsigned short*)carve(M * Cc * 2);
  unsigned short* qb  = (unsigned short*)carve(M * Cc * 2);
  unsigned short* kb  = (unsigned short*)carve(M * Cc * 2);
  unsigned short* vb  = (unsigned short*)carve(M * Cc * 2);
  unsigned short* yb  = (unsigned short*)carve(M * Cc * 2);
  unsigned short* mid = (unsigned short*)carve(M * FF * 2);

  // weight conversion (once per launch; deterministic)
  cvt_f32_bf16<<<2048, 256, 0, stream>>>(Wq, WqB, nQK);
  cvt_f32_bf16<<<2048, 256, 0, stream>>>(Wk, WkB, nQK);
  cvt_f32_bf16<<<2048, 256, 0, stream>>>(Wv, WvB, nQK);
  cvt_f32_bf16<<<2048, 256, 0, stream>>>(Wo, WoB, nQK);
  cvt_f32_bf16<<<2048, 256, 0, stream>>>(W1, W1B, nFF);
  cvt_f32_bf16<<<2048, 256, 0, stream>>>(W2, W2B, nFF);
  cvt_f32_bf16<<<2048, 256, 0, stream>>>(Wh, WhB, (size_t)OUTc * Cc);

  add_pos<<<2048, 256, 0, stream>>>(seq, pos, x, M * Cc, (size_t)Tc * Cc);

  const dim3 gC(Cc / 64, M / 128);      // N=1024 GEMMs
  const dim3 gF(FF / 64, M / 128);      // N=4096 GEMM
  const dim3 gA(Tc / 64, Bc * Hc);      // attention

  for (int l = 0; l < Lc; ++l) {
    const size_t wq = (size_t)l * Cc * Cc, wf = (size_t)l * FF * Cc;
    layernorm_bf16<<<M, 256, 0, stream>>>(x, l1w + l * Cc, l1b + l * Cc, hb, Cc);
    gemm_bf16<true, false, false, false><<<gC, 128, 0, stream>>>(
        hb, WqB + wq, bq + l * Cc, nullptr, nullptr, qb, (int)M, Cc, Cc);
    gemm_bf16<true, false, false, false><<<gC, 128, 0, stream>>>(
        hb, WkB + wq, bk + l * Cc, nullptr, nullptr, kb, (int)M, Cc, Cc);
    gemm_bf16<true, false, false, false><<<gC, 128, 0, stream>>>(
        hb, WvB + wq, bv + l * Cc, nullptr, nullptr, vb, (int)M, Cc, Cc);
    flash_attn<<<gA, 128, 0, stream>>>(qb, kb, vb, yb);
    gemm_bf16<true, false, true, true><<<gC, 128, 0, stream>>>(
        yb, WoB + wq, bo + l * Cc, x, x, nullptr, (int)M, Cc, Cc);
    layernorm_bf16<<<M, 256, 0, stream>>>(x, l2w + l * Cc, l2b + l * Cc, hb, Cc);
    gemm_bf16<true, true, false, false><<<gF, 128, 0, stream>>>(
        hb, W1B + wf, b1 + l * FF, nullptr, nullptr, mid, (int)M, FF, Cc);
    gemm_bf16<true, false, true, true><<<gC, 128, 0, stream>>>(
        mid, W2B + wf, b2 + l * Cc, x, x, nullptr, (int)M, Cc, FF);
  }

  layernorm_bf16<<<M, 256, 0, stream>>>(x, lnfw, lnfb, hb, Cc);
  gemm_bf16<false, false, false, true><<<dim3(OUTc / 64, M / 128), 128, 0, stream>>>(
      hb, WhB, nullptr, nullptr, (float*)d_out, nullptr, (int)M, OUTc, Cc);
}
